// MotifConvolution_10161892622472
// MI455X (gfx1250) — matrix operand order repaired and checked by reading, not compiled, not run on hardware
//
#include <hip/hip_runtime.h>
#include <hip/hip_bf16.h>

#define F 128

typedef __attribute__((ext_vector_type(16))) __bf16 v16bf;
typedef __attribute__((ext_vector_type(16))) float  v16f;
typedef __attribute__((ext_vector_type(8)))  float  v8f;

union BFragU { v16bf v; uint4 q[2]; };

// Pack two f32 -> two bf16 in one DWORD (lowers to v_cvt_pk_bf16_f32).
__device__ __forceinline__ unsigned packbf2(float lo, float hi) {
  union { __bf16 b[2]; unsigned u; } p;
  p.b[0] = (__bf16)lo;
  p.b[1] = (__bf16)hi;
  return p.u;
}

// XW[N x 128] = X[N x 128] * W[128 x 128]; bf16 operands, f32 accumulate.
// 256 threads = 8 waves; wave w computes rows [16*(8*blk+w), +16).
//
// W is staged in LDS in a fragment-native, lane-contiguous layout:
//   dword index = (((kc*8 + nt)*2 + g)*32 + lane)*4 + qq      (j = 4g + qq)
//   holds {W[k][n], W[k+1][n]},  k = kc*32 + (lane>>4)*16 + 2j,  n = nt*16 + (lane&15)
// so each lane fetches a B fragment as two ds_load_b128 into contiguous VGPR
// quads (no register shuffling), with lane L hitting banks 4L..4L+3 -> conflict-free.
__global__ __launch_bounds__(256) void gemm_xw_bf16(
    const float* __restrict__ x, const float* __restrict__ w,
    float* __restrict__ xw, int n_rows) {
  __shared__ unsigned lds_w[8192];   // 32 KB

  const int tid = threadIdx.x;
  for (int e = tid; e < 8192; e += 256) {
    int qq = e & 3;
    int ln = (e >> 2) & 31;
    int g  = (e >> 7) & 1;
    int nt = (e >> 8) & 7;
    int kc = e >> 11;
    int j  = g * 4 + qq;
    int n  = nt * 16 + (ln & 15);
    int k  = kc * 32 + (ln >> 4) * 16 + 2 * j;
    lds_w[e] = packbf2(w[(size_t)k * F + n], w[(size_t)(k + 1) * F + n]);
  }
  __syncthreads();

  const int wave = tid >> 5;
  const int lane = tid & 31;
  const int h    = lane >> 4;   // lane half selects K sub-range
  const int c    = lane & 15;   // A: row within tile; B/C: column within tile

  const int mtile  = blockIdx.x * 8 + wave;
  const int mtiles = n_rows >> 4;
  if (mtile >= mtiles) return;  // wave-uniform: EXEC stays all-1 for WMMA

  const float* xr = x + (size_t)(mtile * 16 + c) * F;

  v8f acc[8] = {};

  #pragma unroll
  for (int kc = 0; kc < 4; ++kc) {
    // A fragment: lane (h,c) holds row mtile*16+c, K = kc*32 + h*16 + {0..15}
    const float* ap = xr + kc * 32 + h * 16;
    float4 a0 = *(const float4*)(ap);
    float4 a1 = *(const float4*)(ap + 4);
    float4 a2 = *(const float4*)(ap + 8);
    float4 a3 = *(const float4*)(ap + 12);
    v16f af32;
    af32[0]  = a0.x; af32[1]  = a0.y; af32[2]  = a0.z; af32[3]  = a0.w;
    af32[4]  = a1.x; af32[5]  = a1.y; af32[6]  = a1.z; af32[7]  = a1.w;
    af32[8]  = a2.x; af32[9]  = a2.y; af32[10] = a2.z; af32[11] = a2.w;
    af32[12] = a3.x; af32[13] = a3.y; af32[14] = a3.z; af32[15] = a3.w;
    v16bf av = __builtin_convertvector(af32, v16bf);

    #pragma unroll
    for (int nt = 0; nt < 8; ++nt) {
      const unsigned* bp = &lds_w[(size_t)((kc * 8 + nt) * 2) * 128 + lane * 4];
      BFragU bf;
      bf.q[0] = *(const uint4*)(bp);
      bf.q[1] = *(const uint4*)(bp + 128);
      acc[nt] = __builtin_amdgcn_wmma_f32_16x16x32_bf16(
          false, av, false, bf.v, (short)0, acc[nt], false, false);
    }
  }

  // C layout: VGPR r of lane (h,c) = C[M = h*8 + r, N = c]
  #pragma unroll
  for (int nt = 0; nt < 8; ++nt) {
    float* orow = xw + (size_t)(mtile * 16 + h * 8) * F + nt * 16 + c;
    #pragma unroll
    for (int r = 0; r < 8; ++r) orow[(size_t)r * F] = acc[nt][r];
  }
}

// One wave32 per edge: gather float4/lane from XW[col], scale, atomic-add into out[row].
// XW (51.2 MB) + motif accumulator (51.2 MB) both fit in the 192 MB L2.
__global__ __launch_bounds__(256) void spmm_scatter(
    const float* __restrict__ xw, const float* __restrict__ val,
    const int* __restrict__ row, const int* __restrict__ col,
    float* __restrict__ out, int nedges) {
  int e = blockIdx.x * 8 + (threadIdx.x >> 5);
  if (e >= nedges) return;
  int lane = threadIdx.x & 31;
  int r = row[e];                 // wave-uniform loads (broadcast)
  int c = col[e];
  float v = val[e];
  const float4 s = *(const float4*)(xw + (size_t)c * F + lane * 4);
  float* d = out + (size_t)r * F + lane * 4;
  atomicAdd(d + 0, v * s.x);
  atomicAdd(d + 1, v * s.y);
  atomicAdd(d + 2, v * s.z);
  atomicAdd(d + 3, v * s.w);
}

__global__ void zero_f32(float* __restrict__ p, size_t n) {
  size_t i = (size_t)blockIdx.x * blockDim.x + threadIdx.x;
  if (i < n) p[i] = 0.0f;
}

__global__ void elu_inplace(float* __restrict__ p, size_t n) {
  size_t i = (size_t)blockIdx.x * blockDim.x + threadIdx.x;
  if (i < n) {
    float xv = p[i];
    p[i] = xv > 0.0f ? xv : (__expf(xv) - 1.0f);
  }
}

extern "C" void kernel_launch(void* const* d_in, const int* in_sizes, int n_in,
                              void* d_out, int out_size, void* d_ws, size_t ws_size,
                              hipStream_t stream) {
  const float* x = (const float*)d_in[0];
  const int n_rows = in_sizes[0] / F;     // N = 100000 (multiple of 16)
  float* out = (float*)d_out;             // [2][N][128] f32
  float* xw  = (float*)d_ws;              // scratch: [N][128] f32 (51.2 MB)

  struct Pos { int w, val, row, col, motif; };
  const Pos pos[5] = {
      {1, 2, 3, 4, 0},  {5, 6, 7, 8, 0},
      {9, 10, 11, 12, 1}, {13, 14, 15, 16, 1}, {17, 18, 19, 20, 1},
  };

  const size_t total = (size_t)2 * n_rows * F;
  zero_f32<<<(unsigned)((total + 255) / 256), 256, 0, stream>>>(out, total);

  const int mtiles = n_rows / 16;
  const int gemm_blocks = (mtiles + 7) / 8;

  for (int p = 0; p < 5; ++p) {
    const float* w   = (const float*)d_in[pos[p].w];
    const float* val = (const float*)d_in[pos[p].val];
    const int*   row = (const int*)d_in[pos[p].row];
    const int*   col = (const int*)d_in[pos[p].col];
    const int E = in_sizes[pos[p].val];
    float* motif_out = out + (size_t)pos[p].motif * n_rows * F;

    gemm_xw_bf16<<<gemm_blocks, 256, 0, stream>>>(x, w, xw, n_rows);
    spmm_scatter<<<(E + 7) / 8, 256, 0, stream>>>(xw, val, row, col, motif_out, E);
  }

  elu_inplace<<<(unsigned)((total + 255) / 256), 256, 0, stream>>>(out, total);
}
